// E_Path_Tracer_15169824489474
// MI455X (gfx1250) — compile-verified, matrix operand order checked by reading, not run on hardware
//
#include <hip/hip_runtime.h>
#include <hip/hip_bf16.h>
#include <cstdint>
#include <cstddef>

// ---------------------------------------------------------------------------
// Problem constants (reference): B=256, H=128, P=10, L=20, PD=64, NH=4
// ---------------------------------------------------------------------------
#define BB   256
#define HH   128
#define PP   10
#define LL   20
#define PD   64
#define H2   256
#define RROWS (BB * PP * LL)   // 51200 rows for the fc MLP (multiple of 16)

typedef __attribute__((ext_vector_type(16))) _Float16 v16h;
typedef __attribute__((ext_vector_type(8)))  _Float16 v8h;
typedef __attribute__((ext_vector_type(8)))  float    v8f;
typedef __attribute__((ext_vector_type(4)))  int      v4i;

#define WMMA_F16(a, b, c) \
  __builtin_amdgcn_wmma_f32_16x16x32_f16(false, (a), false, (b), (short)0, (c), false, false)

// Optional CDNA5 async global->LDS path (probe via __has_builtin; fallback = sync copy)
#if defined(__has_builtin)
#if __has_builtin(__builtin_amdgcn_global_load_async_to_lds_b128)
#define HAVE_ASYNC_G2L 1
#endif
#if __has_builtin(__builtin_amdgcn_s_wait_asynccnt)
#define HAVE_WAIT_ASYNC_BUILTIN 1
#endif
#endif

typedef __attribute__((address_space(1))) v4i* gp_v4i;
typedef __attribute__((address_space(3))) v4i* lp_v4i;

__device__ __forceinline__ float sigmoidf_(float x) { return 1.0f / (1.0f + __expf(-x)); }

// ---------------------------------------------------------------------------
// WMMA 16-bit fragment load (wave32 layout, CDNA5 ISA 7.12.2).
// Both A (row-major MxK) and B (stored as B^T, row-major NxK) use:
//   lane: m/n = lane&15, half = lane>>4
//   ptr = base + (row0|col0 + lane&15)*ld + k0 + half*8 ; read 8 halves @+0, @+16
// -> two 16-byte contiguous loads per lane (global_load_b128 / ds_load_b128).
// ---------------------------------------------------------------------------
__device__ __forceinline__ const _Float16* frag_ptr(const _Float16* base, int rc0,
                                                    int ld, int k0, int lane) {
  return base + (size_t)(rc0 + (lane & 15)) * ld + k0 + ((lane >> 4) << 3);
}

__device__ __forceinline__ v16h load_frag16(const _Float16* p) {
  v8h lo = *(const v8h*)(p);
  v8h hi = *(const v8h*)(p + 16);
  return __builtin_shufflevector(lo, hi, 0, 1, 2, 3, 4, 5, 6, 7,
                                 8, 9, 10, 11, 12, 13, 14, 15);
}

// 16-byte global -> LDS copy; async (ASYNCcnt-tracked) path when available.
__device__ __forceinline__ void copy16_g2l(_Float16* ldst, const _Float16* gsrc) {
#if defined(HAVE_ASYNC_G2L)
  __builtin_amdgcn_global_load_async_to_lds_b128((gp_v4i)(gsrc), (lp_v4i)(ldst), 0, 0);
#else
  *(v8h*)ldst = *(const v8h*)gsrc;
#endif
}

__device__ __forceinline__ void wait_async_g2l() {
#if defined(HAVE_ASYNC_G2L)
#if defined(HAVE_WAIT_ASYNC_BUILTIN)
  __builtin_amdgcn_s_wait_asynccnt(0);
#else
  asm volatile("s_wait_asynccnt 0" ::: "memory");
#endif
#endif
}

// ---------------------------------------------------------------------------
// K0: fused f32->f16 convert + transpose (LDS tiled, coalesced both sides).
// src: [batch][K][M] f32 ; dst: [batch][M][K] f16
// ---------------------------------------------------------------------------
__global__ void k_convert_transpose(const float* __restrict__ src, _Float16* __restrict__ dst,
                                    int K, int M, size_t srcStride, size_t dstStride) {
  __shared__ float tile[32][33];
  const float* s = src + (size_t)blockIdx.z * srcStride;
  _Float16* d = dst + (size_t)blockIdx.z * dstStride;
  int k0 = blockIdx.y * 32, m0 = blockIdx.x * 32;
  int tx = threadIdx.x & 31, ty = threadIdx.x >> 5;  // 32 x 8
  for (int r = ty; r < 32; r += 8) {
    int k = k0 + r, m = m0 + tx;
    tile[r][tx] = (k < K && m < M) ? s[(size_t)k * M + m] : 0.f;
  }
  __syncthreads();
  for (int r = ty; r < 32; r += 8) {
    int m = m0 + r, k = k0 + tx;
    if (m < M && k < K) d[(size_t)m * K + k] = (_Float16)tile[tx][r];
  }
}

// ---------------------------------------------------------------------------
// K1: BatchNorm stats (channel c owns flat [c*200,(c+1)*200) per batch).
// ---------------------------------------------------------------------------
__global__ void k_bn_stats(const float* __restrict__ fr, const float* __restrict__ bk,
                           float* __restrict__ stats) {
  int s = blockIdx.x >> 6, c = blockIdx.x & 63;
  const float* x = (s == 0) ? fr : bk;
  float sum = 0.f, sq = 0.f;
  for (int i = threadIdx.x; i < BB * 200; i += 256) {
    int b = i / 200, j = i - b * 200;
    float v = x[(size_t)b * (PP * LL * PD) + c * 200 + j];
    sum += v; sq += v * v;
  }
  __shared__ float s1[256], s2[256];
  s1[threadIdx.x] = sum; s2[threadIdx.x] = sq;
  __syncthreads();
  for (int off = 128; off > 0; off >>= 1) {
    if (threadIdx.x < off) { s1[threadIdx.x] += s1[threadIdx.x + off];
                             s2[threadIdx.x] += s2[threadIdx.x + off]; }
    __syncthreads();
  }
  if (threadIdx.x == 0) {
    float inv = 1.0f / (float)(BB * 200);
    float m = s1[0] * inv;
    float var = s2[0] * inv - m * m;
    stats[(s * 64 + c) * 2 + 0] = m;
    stats[(s * 64 + c) * 2 + 1] = rsqrtf(var + 1e-5f);
  }
}

// ---------------------------------------------------------------------------
// K2: normalize + emit f16 xn[s][b*200 + p*20 + l][d]
// ---------------------------------------------------------------------------
__global__ void k_bn_norm(const float* __restrict__ fr, const float* __restrict__ bk,
                          const float* __restrict__ stats, const float* __restrict__ gamma,
                          const float* __restrict__ beta, _Float16* __restrict__ xn) {
  size_t idx = (size_t)blockIdx.x * 256 + threadIdx.x;
  const size_t per = (size_t)BB * (PP * LL * PD);
  if (idx >= 2 * per) return;
  int s = (int)(idx / per);
  size_t rem = idx - (size_t)s * per;
  int b = (int)(rem / (PP * LL * PD));
  int j = (int)(rem - (size_t)b * (PP * LL * PD));
  int ch = j / 200;
  const float* x = s ? bk : fr;
  float m = stats[(s * 64 + ch) * 2 + 0];
  float r = stats[(s * 64 + ch) * 2 + 1];
  float v = (x[(size_t)b * (PP * LL * PD) + j] - m) * r * gamma[s * 64 + ch] + beta[s * 64 + ch];
  int row = j >> 6, col = j & 63;
  xn[((size_t)s * RROWS + (size_t)b * (PP * LL) + row) * PD + col] = (_Float16)v;
}

// ---------------------------------------------------------------------------
// K3: ker[s][b][k] = (sign(cat(h_global, hs)) > 0) ? 1 : 0  as f16 (exact)
// ---------------------------------------------------------------------------
__global__ void k_ker(const float* __restrict__ hg, const float* __restrict__ frh,
                      const float* __restrict__ bkh, _Float16* __restrict__ ker) {
  int idx = blockIdx.x * 256 + threadIdx.x;
  if (idx >= 2 * BB * H2) return;
  int s = idx / (BB * H2);
  int rem = idx - s * BB * H2;
  int b = rem >> 8, k = rem & 255;
  float v = (k < HH) ? hg[b * HH + k] : (s ? bkh : frh)[b * HH + (k - HH)];
  ker[idx] = (_Float16)((v > 0.f) ? 1.f : 0.f);
}

// ---------------------------------------------------------------------------
// K4: fc GEMM  Out[r,n] = tanh(A[r,:K] @ W[:K,n] + bias[n]); W given as W^T [N][K].
// grid (R/16, 2 streams), block 256 = 8 waves, wave w owns N-tile w. N=128.
// ---------------------------------------------------------------------------
__global__ void k_fc_gemm_tanh(const _Float16* __restrict__ A, size_t aStride, int K,
                               const _Float16* __restrict__ Bt, size_t bStride,
                               const float* __restrict__ bias, size_t biasStride,
                               _Float16* __restrict__ Out, size_t outStride) {
  int s = blockIdx.y;
  const _Float16* As = A + (size_t)s * aStride;
  const _Float16* Bs = Bt + (size_t)s * bStride;
  const float* bs = bias + (size_t)s * biasStride;
  _Float16* Os = Out + (size_t)s * outStride;
  int wave = threadIdx.x >> 5, lane = threadIdx.x & 31;
  int row0 = blockIdx.x << 4;
  int col0 = wave << 4;
  v8f acc = {};
  for (int k0 = 0; k0 < K; k0 += 32) {
    v16h a = load_frag16(frag_ptr(As, row0, K, k0, lane));
    v16h b = load_frag16(frag_ptr(Bs, col0, K, k0, lane));
    acc = WMMA_F16(a, b, acc);
  }
  int n = lane & 15, mb = (lane >> 4) << 3;
  float bv = bs[col0 + n];
#pragma unroll
  for (int r = 0; r < 8; ++r) {
    int row = row0 + mb + r;
    Os[(size_t)row * HH + col0 + n] = (_Float16)tanhf(acc[r] + bv);
  }
}

// ---------------------------------------------------------------------------
// K5: hypernetwork GEMM (dominant).  B operand = hypW16T[g] : [16384][256] f16.
// Output stored TRANSPOSED per (s,b,g) matrix:  WhypT[..][j][i] = W[i][j]
// (column-major W) so the LSTM's B fragments are contiguous.
// grid (16384/128, 8 g, 2*16), block 256 (8 waves).
// ---------------------------------------------------------------------------
__global__ void k_hyper_gemm(const _Float16* __restrict__ ker16,
                             const _Float16* __restrict__ hypW16T,
                             const float* __restrict__ hyp_b,
                             _Float16* __restrict__ WhypT) {
  int s = blockIdx.z >> 4;
  int rowTile = blockIdx.z & 15;
  int g = blockIdx.y;
  int wave = threadIdx.x >> 5, lane = threadIdx.x & 31;
  int col0 = blockIdx.x * 128 + (wave << 4);       // m index 0..16383
  const _Float16* A = ker16 + (size_t)s * H2 * H2; // [256 x 256]
  const _Float16* Bm = hypW16T + (size_t)g * (size_t)(HH * HH) * H2;
  // keep the streaming B operand warm in L2 (global_prefetch_b8)
  __builtin_prefetch(Bm + (size_t)(col0 + 16 + (lane & 15)) * H2, 0, 1);
  v8f acc = {};
#pragma unroll
  for (int kc = 0; kc < 8; ++kc) {
    v16h a = load_frag16(frag_ptr(A, rowTile * 16, H2, kc * 32, lane));
    v16h b = load_frag16(frag_ptr(Bm, col0, H2, kc * 32, lane));
    acc = WMMA_F16(a, b, acc);
  }
  int n = lane & 15, mb = (lane >> 4) << 3;
  int m = col0 + n;
  int jj = m & 127;   // W column (k)
  int ii = m >> 7;    // W row (h, contraction)
  float bv = hyp_b[(size_t)g * (HH * HH) + m];
#pragma unroll
  for (int r = 0; r < 8; ++r) {
    int brow = rowTile * 16 + mb + r;
    float v = tanhf(acc[r] + bv);
    WhypT[(((size_t)(s * BB + brow) * 8 + g) * (HH * HH)) + (size_t)jj * HH + ii] = (_Float16)v;
  }
}

// ---------------------------------------------------------------------------
// K6: hypernetwork biases: bg[s][b][q][u] = tanh(ker . hypb_W[q][:,u] + hypb_b[q][u])
// ---------------------------------------------------------------------------
__global__ void k_bg(const _Float16* __restrict__ ker16, const float* __restrict__ hypbW,
                     const float* __restrict__ hypbB, float* __restrict__ bg) {
  int idx = blockIdx.x * 256 + threadIdx.x;
  if (idx >= 2 * BB * 4 * HH) return;
  int s = idx / (BB * 4 * HH);
  int rem = idx - s * (BB * 4 * HH);
  int b = rem / (4 * HH);
  int q = (rem >> 7) & 3;
  int u = rem & 127;
  const _Float16* kr = ker16 + (size_t)(s * BB + b) * H2;
  float acc = hypbB[q * HH + u];
  for (int k = 0; k < H2; ++k)
    acc += (float)kr[k] * hypbW[((size_t)q * H2 + k) * HH + u];
  bg[idx] = tanhf(acc);
}

// ---------------------------------------------------------------------------
// K7: per-(stream,batch) 20-step LSTM scan.
// One workgroup per (s,b), 8 waves, wave w owns output cols [16w,16w+16).
// The 8 transposed weight matrices (256 KB) are staged ONCE into LDS with
// async global->LDS b128 copies (ASYNCcnt), padded to 136-half rows to spread
// LDS banks; xt/h tiles live in LDS; c state lives in 8 registers per lane.
// Gate matrices: 0 Wii 1 Whi 2 Wif 3 Whf 4 Wig 5 Whg 6 Wio 7 Who.
// ---------------------------------------------------------------------------
#define WROW 136            // padded row stride (halves): 272 B -> 4-bank skew
#define WMAT (128 * WROW)   // halves per staged matrix

__global__ void k_lstm_scan(const _Float16* __restrict__ x2,
                            const _Float16* __restrict__ WhypT,
                            const float* __restrict__ bg,
                            float* __restrict__ hp) {
  extern __shared__ __align__(16) char smem[];
  _Float16* Wl = (_Float16*)smem;            // 8 * 128 * WROW halves
  _Float16* xt = Wl + 8 * WMAT;              // 16 * WROW
  _Float16* hb = xt + 16 * WROW;             // 16 * WROW

  int s = blockIdx.y, b = blockIdx.x;
  int tid = threadIdx.x;
  // zero xt/hb (incl. pad columns; pad stays zero forever)
  for (int i = tid; i < 16 * WROW; i += 256) { xt[i] = (_Float16)0.f; hb[i] = (_Float16)0.f; }

  // stage the 8 column-major weight matrices: 131072 halves = 16384 x 16B chunks
  const _Float16* Wsb = WhypT + (size_t)(s * BB + b) * 8 * (HH * HH);
  for (int c = tid; c < 16384; c += 256) {
    int g = c >> 11;            // 2048 chunks per matrix
    int cm = c & 2047;
    int j = cm >> 4;            // row (W column index) 0..127
    int off = (cm & 15) << 3;   // halves within row
    copy16_g2l(Wl + (size_t)g * WMAT + j * WROW + off, Wsb + (size_t)c * 8);
  }
  wait_async_g2l();
  __syncthreads();

  const _Float16* x2s = x2 + (size_t)s * RROWS * HH;
  const float* bgs = bg + ((size_t)(s * BB + b) * 4) * HH;
  int wave = tid >> 5, lane = tid & 31;
  int col0 = wave << 4;
  int n = lane & 15, mb = (lane >> 4) << 3;
  v8f cstate = {};   // c for (M = mb+r, N = col0+n), fixed lane ownership

  for (int t = 0; t < LL; ++t) {
    // stage x_t (10 real rows; rows 10..15 and pad cols remain zero)
    for (int c = tid; c < PP * 16; c += 256) {   // 160 x 16B chunks
      int p = c >> 4, off = (c & 15) << 3;
      *(v8h*)(xt + p * WROW + off) =
          *(const v8h*)(x2s + ((size_t)b * (PP * LL) + p * LL + t) * HH + off);
    }
    __syncthreads();

    v8f acc0 = {}, acc1 = {}, acc2 = {}, acc3 = {};
#pragma unroll
    for (int kc = 0; kc < 4; ++kc) {
      int k0 = kc * 32;
      v16h ax = load_frag16(frag_ptr(xt, 0, WROW, k0, lane));
      v16h ah = load_frag16(frag_ptr(hb, 0, WROW, k0, lane));
      acc0 = WMMA_F16(ax, load_frag16(frag_ptr(Wl + 0 * WMAT, col0, WROW, k0, lane)), acc0);
      acc0 = WMMA_F16(ah, load_frag16(frag_ptr(Wl + 1 * WMAT, col0, WROW, k0, lane)), acc0);
      acc1 = WMMA_F16(ax, load_frag16(frag_ptr(Wl + 2 * WMAT, col0, WROW, k0, lane)), acc1);
      acc1 = WMMA_F16(ah, load_frag16(frag_ptr(Wl + 3 * WMAT, col0, WROW, k0, lane)), acc1);
      acc2 = WMMA_F16(ax, load_frag16(frag_ptr(Wl + 4 * WMAT, col0, WROW, k0, lane)), acc2);
      acc2 = WMMA_F16(ah, load_frag16(frag_ptr(Wl + 5 * WMAT, col0, WROW, k0, lane)), acc2);
      acc3 = WMMA_F16(ax, load_frag16(frag_ptr(Wl + 6 * WMAT, col0, WROW, k0, lane)), acc3);
      acc3 = WMMA_F16(ah, load_frag16(frag_ptr(Wl + 7 * WMAT, col0, WROW, k0, lane)), acc3);
    }
    __syncthreads();  // all waves done reading hb before it is rewritten

#pragma unroll
    for (int r = 0; r < 8; ++r) {
      int M = mb + r, C = col0 + n;
      float iv = sigmoidf_(acc0[r] + bgs[0 * HH + C]);
      float fv = sigmoidf_(acc1[r] + bgs[1 * HH + C]);
      float gv = tanhf    (acc2[r] + bgs[2 * HH + C]);
      float ov = sigmoidf_(acc3[r] + bgs[3 * HH + C]);
      float cNew = fv * cstate[r] + iv * gv;
      cstate[r] = cNew;
      hb[M * WROW + C] = (_Float16)(ov * tanhf(cNew));
    }
    // next iteration's xt staging + barrier publishes hb before the next WMMA reads
  }
  __syncthreads();
  for (int i = tid; i < PP * HH; i += 256)
    hp[((size_t)(s * BB + b) * PP) * HH + i] = (float)hb[(i >> 7) * WROW + (i & 127)];
}

// ---------------------------------------------------------------------------
// K8: multi-head attention over paths + global LSTMCell. One block per (s,b).
// ---------------------------------------------------------------------------
__global__ void k_attn_cell(const float* __restrict__ hp, const float* __restrict__ h_global,
                            const float* __restrict__ pre_att_W, const float* __restrict__ pre_att_b,
                            const float* __restrict__ att_W, const float* __restrict__ att_b,
                            const float* __restrict__ fr_h, const float* __restrict__ fr_c,
                            const float* __restrict__ bk_h, const float* __restrict__ bk_c,
                            const float* __restrict__ Wih, const float* __restrict__ Whh,
                            const float* __restrict__ bih, const float* __restrict__ bhh,
                            float* __restrict__ out) {
  int s = blockIdx.y, b = blockIdx.x, tid = threadIdx.x;  // 128 threads
  __shared__ float hpL[PP][HH], hg[HH], hsL[HH], sc[PP][4], aw[PP][4], attL[HH];
  const float* hpb = hp + (size_t)(s * BB + b) * PP * HH;
  for (int i = tid; i < PP * HH; i += 128) hpL[i >> 7][i & 127] = hpb[i];
  hg[tid] = h_global[b * HH + tid];
  const float* hs = (s == 0) ? fr_h : bk_h;
  const float* cs = (s == 0) ? fr_c : bk_c;
  hsL[tid] = hs[b * HH + tid];
  __syncthreads();

  if (tid < PP * 4) {  // scores: cat(h_global, hp[p]) . att_W[s][n]
    int p = tid >> 2, nh = tid & 3;
    const float* w = att_W + ((size_t)s * 4 + nh) * H2;
    float acc = att_b[s * 4 + nh];
    for (int k = 0; k < HH; ++k) acc += hg[k] * w[k];
    for (int k = 0; k < HH; ++k) acc += hpL[p][k] * w[HH + k];
    sc[p][nh] = acc;
  }
  __syncthreads();
  if (tid < 4) {  // softmax over P
    float mx = -1e30f;
    for (int p = 0; p < PP; ++p) mx = fmaxf(mx, sc[p][tid]);
    float sum = 0.f;
    for (int p = 0; p < PP; ++p) { float e = __expf(sc[p][tid] - mx); aw[p][tid] = e; sum += e; }
    float inv = 1.f / sum;
    for (int p = 0; p < PP; ++p) aw[p][tid] *= inv;
  }
  __syncthreads();
  {  // att[n*32+k] = sum_p a[p,n] * (hp[p] . pre_att_W[s][n][:,k] + pre_att_b)
    int nh = tid >> 5, kk = tid & 31;
    const float* pw = pre_att_W + (((size_t)s * 4 + nh) * HH) * 32;
    float pb = pre_att_b[(s * 4 + nh) * 32 + kk];
    float acc = 0.f;
    for (int p = 0; p < PP; ++p) {
      float v = pb;
      for (int h = 0; h < HH; ++h) v += hpL[p][h] * pw[h * 32 + kk];
      acc += aw[p][nh] * v;
    }
    attL[tid] = acc;
  }
  __syncthreads();
  // global LSTMCell (torch gate order i,f,g,o)
  float gates[4];
#pragma unroll
  for (int q = 0; q < 4; ++q) {
    int u = q * HH + tid;
    const float* wi = Wih + ((size_t)s * 4 * HH + u) * HH;
    const float* wh = Whh + ((size_t)s * 4 * HH + u) * HH;
    float acc = bih[s * 4 * HH + u] + bhh[s * 4 * HH + u];
    for (int k = 0; k < HH; ++k) acc += attL[k] * wi[k] + hsL[k] * wh[k];
    gates[q] = acc;
  }
  float csv = cs[b * HH + tid];
  float cn = sigmoidf_(gates[1]) * csv + sigmoidf_(gates[0]) * tanhf(gates[2]);
  float hn = sigmoidf_(gates[3]) * tanhf(cn);
  out[((size_t)(s * 2 + 0) * BB + b) * HH + tid] = hn;
  out[((size_t)(s * 2 + 1) * BB + b) * HH + tid] = cn;
}

// ---------------------------------------------------------------------------
// Host launcher
// ---------------------------------------------------------------------------
extern "C" void kernel_launch(void* const* d_in, const int* in_sizes, int n_in,
                              void* d_out, int out_size, void* d_ws, size_t ws_size,
                              hipStream_t stream) {
  (void)in_sizes; (void)n_in; (void)out_size;
  const float* h_global = (const float*)d_in[0];
  const float* fr_pf    = (const float*)d_in[1];
  const float* bk_pf    = (const float*)d_in[2];
  const float* fr_h     = (const float*)d_in[3];
  const float* fr_c     = (const float*)d_in[4];
  const float* bk_h     = (const float*)d_in[5];
  const float* bk_c     = (const float*)d_in[6];
  const float* bn_gamma = (const float*)d_in[7];
  const float* bn_beta  = (const float*)d_in[8];
  const float* fc_W1    = (const float*)d_in[9];
  const float* fc_b1    = (const float*)d_in[10];
  const float* fc_W2    = (const float*)d_in[11];
  const float* fc_b2    = (const float*)d_in[12];
  const float* hyp_W    = (const float*)d_in[13];
  const float* hyp_b    = (const float*)d_in[14];
  const float* hypb_W   = (const float*)d_in[15];
  const float* hypb_b   = (const float*)d_in[16];
  const float* pre_att_W = (const float*)d_in[17];
  const float* pre_att_b = (const float*)d_in[18];
  const float* att_W    = (const float*)d_in[19];
  const float* att_b    = (const float*)d_in[20];
  const float* lstm_Wih = (const float*)d_in[21];
  const float* lstm_Whh = (const float*)d_in[22];
  const float* lstm_bih = (const float*)d_in[23];
  const float* lstm_bhh = (const float*)d_in[24];
  float* out = (float*)d_out;

  // -------- workspace carve-out (deterministic) --------
  char* ws = (char*)d_ws;
  size_t cur = 0;
  auto carve = [&](size_t bytes) -> void* {
    void* p = ws + cur;
    cur = (cur + bytes + 255) & ~(size_t)255;
    return p;
  };
  float*    stats   = (float*)   carve((size_t)2 * 64 * 2 * sizeof(float));
  _Float16* xn      = (_Float16*)carve((size_t)2 * RROWS * PD * 2);
  _Float16* y1      = (_Float16*)carve((size_t)2 * RROWS * HH * 2);
  _Float16* x2      = (_Float16*)carve((size_t)2 * RROWS * HH * 2);
  _Float16* ker16   = (_Float16*)carve((size_t)2 * BB * H2 * 2);
  _Float16* W1T     = (_Float16*)carve((size_t)2 * HH * PD * 2);        // [s][N=128][K=64]
  _Float16* W2T     = (_Float16*)carve((size_t)2 * HH * HH * 2);        // [s][N=128][K=128]
  _Float16* hypW16T = (_Float16*)carve((size_t)8 * (HH * HH) * H2 * 2); // [g][16384][256]
  _Float16* WhypT   = (_Float16*)carve((size_t)2 * BB * 8 * HH * HH * 2);
  float*    bg      = (float*)   carve((size_t)2 * BB * 4 * HH * sizeof(float));
  float*    hp      = (float*)   carve((size_t)2 * BB * PP * HH * sizeof(float));
  if (cur > ws_size) return;  // workspace too small: bail deterministically

  // -------- 0) weight pre-transpose + f16 convert --------
  k_convert_transpose<<<dim3(128 / 32, 64 / 32, 2), 256, 0, stream>>>(
      fc_W1, W1T, PD, HH, (size_t)PD * HH, (size_t)HH * PD);
  k_convert_transpose<<<dim3(128 / 32, 128 / 32, 2), 256, 0, stream>>>(
      fc_W2, W2T, HH, HH, (size_t)HH * HH, (size_t)HH * HH);
  k_convert_transpose<<<dim3((HH * HH) / 32, H2 / 32, 8), 256, 0, stream>>>(
      hyp_W, hypW16T, H2, HH * HH, (size_t)H2 * HH * HH, (size_t)HH * HH * H2);

  // -------- 1) BatchNorm stats + normalize --------
  k_bn_stats<<<dim3(2 * 64), 256, 0, stream>>>(fr_pf, bk_pf, stats);
  {
    size_t total = (size_t)2 * BB * (PP * LL * PD);
    k_bn_norm<<<dim3((unsigned)((total + 255) / 256)), 256, 0, stream>>>(
        fr_pf, bk_pf, stats, bn_gamma, bn_beta, xn);
  }

  // -------- 2) ker (binary, exact in f16) --------
  k_ker<<<dim3((2 * BB * H2 + 255) / 256), 256, 0, stream>>>(h_global, fr_h, bk_h, ker16);

  // -------- 3) fc MLP via WMMA --------
  k_fc_gemm_tanh<<<dim3(RROWS / 16, 2), 256, 0, stream>>>(
      xn, (size_t)RROWS * PD, PD, W1T, (size_t)HH * PD,
      fc_b1, (size_t)HH, y1, (size_t)RROWS * HH);
  k_fc_gemm_tanh<<<dim3(RROWS / 16, 2), 256, 0, stream>>>(
      y1, (size_t)RROWS * HH, HH, W2T, (size_t)HH * HH,
      fc_b2, (size_t)HH, x2, (size_t)RROWS * HH);

  // -------- 4) hypernetwork GEMM (dominant, WMMA) + biases --------
  k_hyper_gemm<<<dim3((HH * HH) / 128, 8, 2 * 16), 256, 0, stream>>>(
      ker16, hypW16T, hyp_b, WhypT);
  k_bg<<<dim3((2 * BB * 4 * HH + 255) / 256), 256, 0, stream>>>(ker16, hypb_W, hypb_b, bg);

  // -------- 5) per-batch LSTM scan (WMMA, 287 KB LDS-resident weights) --------
  {
    const int lstmShmem = (8 * WMAT + 2 * 16 * WROW) * 2;  // 287,232 B <= 320 KB
    (void)hipFuncSetAttribute((const void*)k_lstm_scan,
                              hipFuncAttributeMaxDynamicSharedMemorySize, lstmShmem);
    k_lstm_scan<<<dim3(BB, 2), 256, lstmShmem, stream>>>(x2, WhypT, bg, hp);
  }

  // -------- 6) attention + global LSTM cell -> d_out --------
  k_attn_cell<<<dim3(BB, 2), 128, 0, stream>>>(
      hp, h_global, pre_att_W, pre_att_b, att_W, att_b,
      fr_h, fr_c, bk_h, bk_c, lstm_Wih, lstm_Whh, lstm_bih, lstm_bhh, out);
}
